// PrototypeDistance_EU_13245679141411
// MI455X (gfx1250) — matrix operand order repaired and checked by reading, not compiled, not run on hardware
//
#include <hip/hip_runtime.h>
#include <hip/hip_bf16.h>

typedef __attribute__((ext_vector_type(16))) __bf16 v16bf;
typedef __attribute__((ext_vector_type(8)))  __bf16 v8bf;
typedef __attribute__((ext_vector_type(8)))  float  v8f;
typedef __attribute__((ext_vector_type(4)))  float  v4f;

#define N_ROWS 32768
#define M_COLS 2048
#define KDIM   512
#define LOG2E_F 1.4426950408889634f

// d_ws layout:
//   [0, 2MB)       p_hi  : prototypes split-high bf16, WMMA-B tiled layout
//   [2MB, 4MB)     p_lo  : prototypes split-low  bf16, same layout
//   [4MB, 4MB+8KB) psq   : ||p||^2 per prototype (f32)
//
// Tiled B layout: chunk index = (MT*16 + kstep)*32 + lane, 16 bf16 per chunk.
//   MT = m/16 (column tile), lane = (m%16) + 16*((k%16)>=8), elem = k%8 + 8*((k%32)>=16).

__global__ __launch_bounds__(256) void proto_convert(
    const float* __restrict__ proto,
    __bf16* __restrict__ p_hi, __bf16* __restrict__ p_lo,
    float* __restrict__ psq) {
  __shared__ float partial[8];
  const int t  = threadIdx.x;
  const int tg = blockIdx.x * 256 + t;
  const int m  = tg >> 6;            // 64 threads per prototype row
  const int k0 = (tg & 63) * 8;      // 8 contiguous K per thread

  const float* src = proto + (size_t)m * KDIM + k0;
  v4f f0 = *(const v4f*)src;
  v4f f1 = *(const v4f*)(src + 4);

  const int    MT    = m >> 4;
  const int    lane  = (m & 15) + ((k0 & 8) ? 16 : 0);
  const int    kstep = k0 >> 5;
  const int    e0    = (k0 & 16) ? 8 : 0;      // 0 or 8
  const size_t chunk = ((size_t)(MT * 16 + kstep) * 32 + lane) * 16 + e0;

  v8bf hi, lo;
  float ssq = 0.0f;
#pragma unroll
  for (int i = 0; i < 8; ++i) {
    float x = (i < 4) ? f0[i] : f1[i - 4];
    ssq += x * x;
    __bf16 h = (__bf16)x;
    hi[i] = h;
    lo[i] = (__bf16)(x - (float)h);
  }
  *(v8bf*)(p_hi + chunk) = hi;
  *(v8bf*)(p_lo + chunk) = lo;

#pragma unroll
  for (int s = 16; s > 0; s >>= 1) ssq += __shfl_xor(ssq, s, 32);
  if ((t & 31) == 0) partial[t >> 5] = ssq;
  __syncthreads();
  if (t < 4) psq[blockIdx.x * 4 + t] = partial[2 * t] + partial[2 * t + 1];
}

__global__ __launch_bounds__(512) void fused_gemm_softmax(
    const float* __restrict__ feat,
    const __bf16* __restrict__ p_hi, const __bf16* __restrict__ p_lo,
    const float* __restrict__ psq,
    float* __restrict__ quality, int* __restrict__ max_id,
    float* __restrict__ pred) {
  // A operand staging (split bf16), WMMA A layout: [kstep(16)][lane(32)][16]
  __shared__ __align__(32) __bf16 a_hi[16 * 32 * 16];
  __shared__ __align__(32) __bf16 a_lo[16 * 32 * 16];
  __shared__ float    wredm[16][16];
  __shared__ float    wreds[16][16];
  __shared__ unsigned wreda[16][16];
  __shared__ float    rowmax[16];
  __shared__ float    rowinv[16];

  const int tid  = threadIdx.x;
  const int lane = tid & 31;
  const int w    = tid >> 5;          // 16 waves
  const int r0   = blockIdx.x * 16;

  // ---- Stage feat rows into LDS in A-operand layout, split into hi/lo ----
  // lane<16 holds row=lane, K {0..7,16..23}; lane>=16 holds row=lane-16, K {8..15,24..31}
  {
    const int idx   = tid;                          // (kstep, lane), 512 items
    const int ks    = idx >> 5;
    const int ln    = idx & 31;
    const int row   = ln & 15;
    const int kbase = ks * 32 + ((ln & 16) >> 1);   // +8 for upper lanes
    const float* s0 = feat + (size_t)(r0 + row) * KDIM + kbase;
    v4f x0 = *(const v4f*)s0;
    v4f x1 = *(const v4f*)(s0 + 4);
    v4f y0 = *(const v4f*)(s0 + 16);
    v4f y1 = *(const v4f*)(s0 + 20);
    v16bf hv, lv;
#pragma unroll
    for (int i = 0; i < 16; ++i) {
      float x = (i < 4) ? x0[i] : (i < 8) ? x1[i - 4] : (i < 12) ? y0[i - 8] : y1[i - 12];
      __bf16 h = (__bf16)x;
      hv[i] = h;
      lv[i] = (__bf16)(x - (float)h);
    }
    *(v16bf*)(a_hi + idx * 16) = hv;
    *(v16bf*)(a_lo + idx * 16) = lv;
  }
  __syncthreads();

  // ---- GEMM: wave w owns columns [w*128, w*128+128) as 8 16x16 tiles ----
  const int m0 = w * 128;
  v8f acc[8];
  const v8f vzero = {0.f, 0.f, 0.f, 0.f, 0.f, 0.f, 0.f, 0.f};
#pragma unroll
  for (int t = 0; t < 8; ++t) acc[t] = vzero;

  // B chunk (elements): (w*8 + t)*8192 + ks*512 + lane*16 ; flattened u = ks*8 + t
  const size_t bbase = (size_t)(w * 8) * 8192 + (size_t)lane * 16;
  // depth-2 rotating prefetch buffers
  v16bf bh0 = *(const v16bf*)(p_hi + bbase);
  v16bf bl0 = *(const v16bf*)(p_lo + bbase);
  v16bf bh1 = *(const v16bf*)(p_hi + bbase + 8192);
  v16bf bl1 = *(const v16bf*)(p_lo + bbase + 8192);

  for (int ks = 0; ks < 16; ++ks) {                 // K-step of 32
    v16bf ah = *(const v16bf*)(a_hi + (ks * 32 + lane) * 16);
    v16bf al = *(const v16bf*)(a_lo + (ks * 32 + lane) * 16);
#pragma unroll
    for (int t = 0; t < 8; ++t) {
      v16bf bh = (t & 1) ? bh1 : bh0;               // u parity == t parity
      v16bf bl = (t & 1) ? bl1 : bl0;
      // prefetch u+2 (2 tiles / 6 WMMAs ahead, crosses kstep boundary)
      const int u  = ks * 8 + t;
      const int up = u + 2;
      if (up < 128) {
        const size_t o = bbase + (size_t)(up & 7) * 8192 + (size_t)(up >> 3) * 512;
        if (t & 1) { bh1 = *(const v16bf*)(p_hi + o); bl1 = *(const v16bf*)(p_lo + o); }
        else       { bh0 = *(const v16bf*)(p_hi + o); bl0 = *(const v16bf*)(p_lo + o); }
      }
      // split product: a*b ~= alo*bhi + ahi*bhi + ahi*blo (bl consumed last)
      acc[t] = __builtin_amdgcn_wmma_f32_16x16x32_bf16(false, al, false, bh,
                                                       (short)0, acc[t], false, false);
      acc[t] = __builtin_amdgcn_wmma_f32_16x16x32_bf16(false, ah, false, bh,
                                                       (short)0, acc[t], false, false);
      acc[t] = __builtin_amdgcn_wmma_f32_16x16x32_bf16(false, ah, false, bl,
                                                       (short)0, acc[t], false, false);
    }
  }

  // ---- Epilogue: logits = 20*cross - 10*||p||^2  (-10*||f||^2 cancels in softmax)
  // C/D layout: lane L: col = (L&15); VGPR i: row = i + 8*(L>>4)
  const int h  = lane >> 4;
  const int cl = lane & 15;
#pragma unroll
  for (int t = 0; t < 8; ++t) {
    float pq = psq[m0 + t * 16 + cl];
#pragma unroll
    for (int i = 0; i < 8; ++i)
      acc[t][i] = 20.0f * acc[t][i] - 10.0f * pq;
  }

  // ---- Row max: per-lane over tiles, shuffle over 16-lane half, LDS across waves
  float mx[8];
#pragma unroll
  for (int i = 0; i < 8; ++i) mx[i] = -3.4e38f;
#pragma unroll
  for (int t = 0; t < 8; ++t)
#pragma unroll
    for (int i = 0; i < 8; ++i) mx[i] = fmaxf(mx[i], acc[t][i]);
#pragma unroll
  for (int s = 1; s < 16; s <<= 1)
#pragma unroll
    for (int i = 0; i < 8; ++i) mx[i] = fmaxf(mx[i], __shfl_xor(mx[i], s, 16));
  if (cl == 0)
#pragma unroll
    for (int i = 0; i < 8; ++i) wredm[w][h * 8 + i] = mx[i];
  __syncthreads();
  if (tid < 16) {
    float M = wredm[0][tid];
#pragma unroll
    for (int ww = 1; ww < 16; ++ww) M = fmaxf(M, wredm[ww][tid]);
    rowmax[tid] = M;
  }
  __syncthreads();

  // ---- Row sum of exp + argmax (first index on ties) ----
  float Mv[8];
#pragma unroll
  for (int i = 0; i < 8; ++i) Mv[i] = rowmax[h * 8 + i];
  float sv[8];
  unsigned av[8];
#pragma unroll
  for (int i = 0; i < 8; ++i) { sv[i] = 0.0f; av[i] = 0xFFFFFFFFu; }
#pragma unroll
  for (int t = 0; t < 8; ++t) {
    unsigned col = (unsigned)(m0 + t * 16 + cl);
#pragma unroll
    for (int i = 0; i < 8; ++i) {
      float x = acc[t][i];
      sv[i] += exp2f((x - Mv[i]) * LOG2E_F);
      if (x == Mv[i] && col < av[i]) av[i] = col;
    }
  }
#pragma unroll
  for (int s = 1; s < 16; s <<= 1)
#pragma unroll
    for (int i = 0; i < 8; ++i) {
      sv[i] += __shfl_xor(sv[i], s, 16);
      unsigned o = __shfl_xor(av[i], s, 16);
      if (o < av[i]) av[i] = o;
    }
  if (cl == 0)
#pragma unroll
    for (int i = 0; i < 8; ++i) { wreds[w][h * 8 + i] = sv[i]; wreda[w][h * 8 + i] = av[i]; }
  __syncthreads();
  if (tid < 16) {
    float    s = 0.0f;
    unsigned a = 0xFFFFFFFFu;
#pragma unroll
    for (int ww = 0; ww < 16; ++ww) {
      s += wreds[ww][tid];
      if (wreda[ww][tid] < a) a = wreda[ww][tid];
    }
    float inv = 1.0f / s;
    rowinv[tid] = inv;
    quality[r0 + tid] = inv;          // max prob = exp(M-M)/sum = 1/sum
    max_id[r0 + tid]  = (int)a;
  }
  __syncthreads();

  // ---- Write normalized predictions straight from registers ----
  float Iv[8];
#pragma unroll
  for (int i = 0; i < 8; ++i) Iv[i] = rowinv[h * 8 + i];
#pragma unroll
  for (int t = 0; t < 8; ++t) {
    const size_t cbase = (size_t)(m0 + t * 16 + cl);
#pragma unroll
    for (int i = 0; i < 8; ++i) {
      float p = exp2f((acc[t][i] - Mv[i]) * LOG2E_F) * Iv[i];
      pred[(size_t)(r0 + h * 8 + i) * M_COLS + cbase] = p;
    }
  }
}

extern "C" void kernel_launch(void* const* d_in, const int* in_sizes, int n_in,
                              void* d_out, int out_size, void* d_ws, size_t ws_size,
                              hipStream_t stream) {
  (void)in_sizes; (void)n_in; (void)out_size; (void)ws_size;
  const float* feat  = (const float*)d_in[0];
  const float* proto = (const float*)d_in[1];

  __bf16* p_hi = (__bf16*)d_ws;
  __bf16* p_lo = (__bf16*)((char*)d_ws + (size_t)2 * 1024 * 1024);
  float*  psq  = (float*)((char*)d_ws + (size_t)4 * 1024 * 1024);

  float* quality = (float*)d_out;
  int*   ids     = (int*)d_out + N_ROWS;
  float* pred    = (float*)d_out + 2 * (size_t)N_ROWS;

  proto_convert<<<512, 256, 0, stream>>>(proto, p_hi, p_lo, psq);
  fused_gemm_softmax<<<2048, 512, 0, stream>>>(feat, p_hi, p_lo, psq,
                                               quality, ids, pred);
}
